// downsample_16784732193438
// MI455X (gfx1250) — matrix operand order repaired and checked by reading, not compiled, not run on hardware
//
#include <hip/hip_runtime.h>

// ---------------- constants -------------------------------------------------
constexpr int  B      = 4;
constexpr int  C      = 27;
constexpr int  HW     = 200704;            // 448*448
constexpr int  CHWn   = 5419008;           // 27*448*448
constexpr int  NPOS   = 196;               // patch positions (14*14)
constexpr int  NP     = 224;               // padded to 7*32 (also 14*16)
constexpr int  DP     = 27648;             // 27*1024  (patch attention K)
constexpr int  GGN    = 3136;              // 56*56    (global positions)
constexpr int  GPAD   = 32;                // padded channel dim (27 -> 32)

// ---------------- bf16 helpers ---------------------------------------------
typedef __attribute__((ext_vector_type(16))) __bf16 v16bf;
typedef __attribute__((ext_vector_type(8)))  float  v8f;

__device__ __forceinline__ __bf16 to_bf16(float f) {
  unsigned u; __builtin_memcpy(&u, &f, 4);
  unsigned r = u + 0x7FFFu + ((u >> 16) & 1u);       // round-to-nearest-even
  unsigned short s = (unsigned short)(r >> 16);
  __bf16 b; __builtin_memcpy(&b, &s, 2);
  return b;
}
__device__ __forceinline__ float bf16_to_f(__bf16 h) {
  unsigned short s; __builtin_memcpy(&s, &h, 2);
  unsigned u = ((unsigned)s) << 16;
  float f; __builtin_memcpy(&f, &u, 4);
  return f;
}

union TileU { uint4 u[2]; v16bf v; };

// A operand (M x K tile, row-major source, lda elements):
//   lane<16 : row=lane,    K = {0..7, 16..23}
//   lane>=16: row=lane-16, K = {8..15, 24..31}      (ISA 7.12.2, 16-bit A)
__device__ __forceinline__ v16bf load_a16(const __bf16* base, int lda, int lane) {
  const int mrow  = lane & 15;
  const int khalf = (lane & 16) >> 1;                 // 0 or 8
  const __bf16* p = base + (size_t)mrow * lda + khalf;
  TileU t;
  t.u[0] = *reinterpret_cast<const uint4*>(p);        // K khalf..khalf+7
  t.u[1] = *reinterpret_cast<const uint4*>(p + 16);   // K khalf+16..khalf+23
  return t.v;
}
// B operand (K x N tile, row-major source, ldb elements): lane%16 = N col, elems = K {khalf..khalf+7, khalf+16..khalf+23}
__device__ __forceinline__ v16bf load_b16(const __bf16* base, int ldb, int lane) {
  const int khalf = (lane & 16) >> 1;
  const __bf16* p = base + (size_t)khalf * ldb + (lane & 15);
  v16bf v;
#pragma unroll
  for (int s = 0; s < 8; ++s) {
    v[s]     = p[(size_t)s * ldb];
    v[8 + s] = p[(size_t)(16 + s) * ldb];
  }
  return v;
}
__device__ __forceinline__ v8f wmma_bf16(v16bf a, v16bf b, v8f c) {
  return __builtin_amdgcn_wmma_f32_16x16x32_bf16(false, a, false, b, (short)0, c,
                                                 false, false);
}

// ---------------- kernels ---------------------------------------------------
__global__ void zero_kernel(uint4* p, size_t n16) {
  size_t i = (size_t)blockIdx.x * blockDim.x + threadIdx.x;
  size_t stride = (size_t)gridDim.x * blockDim.x;
  uint4 z; z.x = z.y = z.z = z.w = 0u;
  for (; i < n16; i += stride) p[i] = z;
}

__global__ void gn_stats_kernel(const float* __restrict__ x, float* __restrict__ stats) {
  const int b = blockIdx.x;
  const int tid = threadIdx.x;
  const float* xb = x + (size_t)b * CHWn;
  float s = 0.f, s2 = 0.f;
  for (int i = tid; i < CHWn; i += 256) { float v = xb[i]; s += v; s2 += v * v; }
  __shared__ float r1[256], r2[256];
  r1[tid] = s; r2[tid] = s2; __syncthreads();
  for (int off = 128; off > 0; off >>= 1) {
    if (tid < off) { r1[tid] += r1[tid + off]; r2[tid] += r2[tid + off]; }
    __syncthreads();
  }
  if (tid == 0) {
    float mu  = r1[0] / (float)CHWn;
    float var = r2[0] / (float)CHWn - mu * mu;
    stats[2 * b]     = mu;
    stats[2 * b + 1] = rsqrtf(var + 1e-5f);
  }
}

// GroupNorm + q/k/v 1x1 convs; writes Qt/Vt transposed (pos, d) and Kb (d, pos)
__global__ void qkv_kernel(const float* __restrict__ x,
                           const float* __restrict__ gw, const float* __restrict__ gb,
                           const float* __restrict__ qw, const float* __restrict__ qb,
                           const float* __restrict__ kw, const float* __restrict__ kb,
                           const float* __restrict__ vw, const float* __restrict__ vb,
                           const float* __restrict__ stats,
                           __bf16* __restrict__ Qt, __bf16* __restrict__ Kb,
                           __bf16* __restrict__ Vt) {
  __shared__ float s_qw[C * C], s_kw[C * C], s_vw[C * C];
  __shared__ float s_qb[C], s_kb[C], s_vb[C], s_gw[C], s_gb[C];
  for (int i = threadIdx.x; i < C * C; i += blockDim.x) {
    s_qw[i] = qw[i]; s_kw[i] = kw[i]; s_vw[i] = vw[i];
  }
  if (threadIdx.x < C) {
    s_qb[threadIdx.x] = qb[threadIdx.x]; s_kb[threadIdx.x] = kb[threadIdx.x];
    s_vb[threadIdx.x] = vb[threadIdx.x];
    s_gw[threadIdx.x] = gw[threadIdx.x]; s_gb[threadIdx.x] = gb[threadIdx.x];
  }
  __syncthreads();
  const int b  = blockIdx.y;
  const int hw = blockIdx.x * 256 + threadIdx.x;
  const float mu = stats[2 * b], rstd = stats[2 * b + 1];
  const float* xb = x + (size_t)b * CHWn;
  float xn[C];
#pragma unroll
  for (int c = 0; c < C; ++c)
    xn[c] = (xb[(size_t)c * HW + hw] - mu) * rstd * s_gw[c] + s_gb[c];
  const int head = hw / NPOS, pos = hw % NPOS;
  __bf16* Qb = Qt + (size_t)b * NP * DP + (size_t)pos * DP;
  __bf16* Vb = Vt + (size_t)b * NP * DP + (size_t)pos * DP;
  __bf16* Kbb = Kb + (size_t)b * DP * NP;
  for (int o = 0; o < C; ++o) {
    float q = s_qb[o], k = s_kb[o], v = s_vb[o];
#pragma unroll
    for (int c = 0; c < C; ++c) {
      q += s_qw[o * C + c] * xn[c];
      k += s_kw[o * C + c] * xn[c];
      v += s_vw[o * C + c] * xn[c];
    }
    const int dd = o * 1024 + head;
    Qb[dd] = to_bf16(q);
    Vb[dd] = to_bf16(v);
    Kbb[(size_t)dd * NP + pos] = to_bf16(k);
  }
}

// 8x8 block-mean pooling -> qgT (g,c), kg (c,g), vgT (g,c)
__global__ void pool_kernel(const __bf16* __restrict__ Qt, const __bf16* __restrict__ Kb,
                            const __bf16* __restrict__ Vt,
                            __bf16* __restrict__ qgT, __bf16* __restrict__ kg,
                            __bf16* __restrict__ vgT) {
  const int idx = blockIdx.x * 256 + threadIdx.x;      // B*C*GGN exact
  const int g = idx % GGN;
  const int t = idx / GGN;
  const int c = t % C;
  const int b = t / C;
  const int gy = g / 56, gx = g % 56;
  const __bf16* Qb = Qt + (size_t)b * NP * DP;
  const __bf16* Vb = Vt + (size_t)b * NP * DP;
  const __bf16* Kbb = Kb + (size_t)b * DP * NP;
  float sq = 0.f, sk = 0.f, sv = 0.f;
  for (int dy = 0; dy < 8; ++dy)
    for (int dx = 0; dx < 8; ++dx) {
      const int hw = (gy * 8 + dy) * 448 + gx * 8 + dx;
      const int dd = c * 1024 + hw / NPOS, pos = hw % NPOS;
      sq += bf16_to_f(Qb[(size_t)pos * DP + dd]);
      sv += bf16_to_f(Vb[(size_t)pos * DP + dd]);
      sk += bf16_to_f(Kbb[(size_t)dd * NP + pos]);
    }
  const float inv = 1.f / 64.f;
  qgT[((size_t)b * GGN + g) * GPAD + c] = to_bf16(sq * inv);
  vgT[((size_t)b * GGN + g) * GPAD + c] = to_bf16(sv * inv);
  kg[(size_t)b * GPAD * GGN + (size_t)c * GGN + g] = to_bf16(sk * inv);
}

// S[i,j] = scale * sum_d Q[d,i] K[d,j]   (A = Qt, B = Kb, K-loop 27648)
// 2x2 register blocking: one wave computes a 32x32 output block.
__global__ void patch_gram_kernel(const __bf16* __restrict__ Qt,
                                  const __bf16* __restrict__ Kb,
                                  float* __restrict__ Sp) {
  const int lane = threadIdx.x;
  const int i0 = blockIdx.x * 32, j0 = blockIdx.y * 32, b = blockIdx.z;
  const __bf16* A0 = Qt + (size_t)b * NP * DP + (size_t)i0 * DP;
  const __bf16* A1 = A0 + (size_t)16 * DP;
  const __bf16* B0 = Kb + (size_t)b * DP * NP + j0;
  v8f acc00 = {0.f,0.f,0.f,0.f,0.f,0.f,0.f,0.f};
  v8f acc01 = acc00, acc10 = acc00, acc11 = acc00;
  for (int k0 = 0; k0 < DP; k0 += 32) {
    v16bf a0 = load_a16(A0 + k0, DP, lane);
    v16bf a1 = load_a16(A1 + k0, DP, lane);
    v16bf b0 = load_b16(B0 + (size_t)k0 * NP,      NP, lane);
    v16bf b1 = load_b16(B0 + (size_t)k0 * NP + 16, NP, lane);
    acc00 = wmma_bf16(a0, b0, acc00);
    acc01 = wmma_bf16(a0, b1, acc01);
    acc10 = wmma_bf16(a1, b0, acc10);
    acc11 = wmma_bf16(a1, b1, acc11);
  }
  const float scale = rsqrtf((float)DP);
  const int ncol = lane & 15, mbase = (lane >> 4) * 8;
  float* S = Sp + (size_t)b * NP * NP;
#pragma unroll
  for (int r = 0; r < 8; ++r) {
    const size_t r0 = (size_t)(i0 + mbase + r) * NP;
    const size_t r1 = (size_t)(i0 + 16 + mbase + r) * NP;
    S[r0 + j0 + ncol]      = acc00[r] * scale;
    S[r0 + j0 + 16 + ncol] = acc01[r] * scale;
    S[r1 + j0 + ncol]      = acc10[r] * scale;
    S[r1 + j0 + 16 + ncol] = acc11[r] * scale;
  }
}

__global__ void patch_softmax_kernel(const float* __restrict__ Sp,
                                     __bf16* __restrict__ Wp) {
  const int row = blockIdx.x, b = blockIdx.y, tid = threadIdx.x;
  __bf16* wrow = Wp + ((size_t)b * NP + row) * NP;
  if (row >= NPOS) {                       // uniform branch per block
    if (tid < NP) wrow[tid] = to_bf16(0.f);
    return;
  }
  const float* srow = Sp + ((size_t)b * NP + row) * NP;
  __shared__ float red[256];
  float v = (tid < NPOS) ? srow[tid] : -3.402823e38f;
  red[tid] = v; __syncthreads();
  for (int off = 128; off > 0; off >>= 1) {
    if (tid < off) red[tid] = fmaxf(red[tid], red[tid + off]);
    __syncthreads();
  }
  const float m = red[0]; __syncthreads();
  float e = (tid < NPOS) ? __expf(v - m) : 0.f;
  red[tid] = e; __syncthreads();
  for (int off = 128; off > 0; off >>= 1) {
    if (tid < off) red[tid] += red[tid + off];
    __syncthreads();
  }
  const float inv = 1.f / red[0];
  if (tid < NP) wrow[tid] = to_bf16((tid < NPOS) ? e * inv : 0.f);
}

// H^T[i,d] = sum_j Wp[i,j] Vt[j,d];  store 0.75*H into HIN (b,c,h,w) layout
// 2(i) x 2(dd) register blocking per wave.
__global__ void patch_apply_kernel(const __bf16* __restrict__ Wp,
                                   const __bf16* __restrict__ Vt,
                                   float* __restrict__ HIN) {
  const int lane = threadIdx.x & 31;
  const int wv   = threadIdx.x >> 5;
  const int i0   = blockIdx.x * 32;                // 7 blocks cover 224 rows
  const int dd0  = (blockIdx.y * 4 + wv) * 32;     // 864 dd-groups of 32
  const int b    = blockIdx.z;
  const __bf16* A0 = Wp + (size_t)b * NP * NP + (size_t)i0 * NP;
  const __bf16* A1 = A0 + (size_t)16 * NP;
  const __bf16* B0 = Vt + (size_t)b * NP * DP + dd0;
  v8f acc00 = {0.f,0.f,0.f,0.f,0.f,0.f,0.f,0.f};
  v8f acc01 = acc00, acc10 = acc00, acc11 = acc00;
#pragma unroll
  for (int kj = 0; kj < 7; ++kj) {                 // K = 224
    const int k0 = kj * 32;
    v16bf a0 = load_a16(A0 + k0, NP, lane);
    v16bf a1 = load_a16(A1 + k0, NP, lane);
    v16bf b0 = load_b16(B0 + (size_t)k0 * DP,      DP, lane);
    v16bf b1 = load_b16(B0 + (size_t)k0 * DP + 16, DP, lane);
    acc00 = wmma_bf16(a0, b0, acc00);
    acc01 = wmma_bf16(a0, b1, acc01);
    acc10 = wmma_bf16(a1, b0, acc10);
    acc11 = wmma_bf16(a1, b1, acc11);
  }
  const int ncol = lane & 15, mbase = (lane >> 4) * 8;
  float* H0 = HIN + (size_t)b * CHWn + (size_t)(dd0 + ncol) * NPOS;
  float* H1 = H0 + (size_t)16 * NPOS;
#pragma unroll
  for (int r = 0; r < 8; ++r) {
    const int ia = i0 + mbase + r;
    const int ib = ia + 16;
    if (ia < NPOS) { H0[ia] = 0.75f * acc00[r]; H1[ia] = 0.75f * acc01[r]; }
    if (ib < NPOS) { H0[ib] = 0.75f * acc10[r]; H1[ib] = 0.75f * acc11[r]; }
  }
}

// Sg[i,j] = scale * sum_c qg[c,i] kg[c,j]  (K = 32 padded, single WMMA / tile;
// A tile loaded once per wave and reused across all 196 j-tiles)
__global__ void global_gram_kernel(const __bf16* __restrict__ qgT,
                                   const __bf16* __restrict__ kg,
                                   float* __restrict__ Sg, int b) {
  const int lane = threadIdx.x;
  const int i0 = blockIdx.x * 16;
  v16bf a = load_a16(qgT + ((size_t)b * GGN + i0) * GPAD, GPAD, lane);
  const __bf16* Bb = kg + (size_t)b * GPAD * GGN;
  const float scale = rsqrtf(27.f);
  const int ncol = lane & 15, mbase = (lane >> 4) * 8;
  for (int j0 = 0; j0 < GGN; j0 += 16) {
    v8f acc = {0.f,0.f,0.f,0.f,0.f,0.f,0.f,0.f};
    v16bf bm = load_b16(Bb + j0, GGN, lane);
    acc = wmma_bf16(a, bm, acc);
#pragma unroll
    for (int r = 0; r < 8; ++r)
      Sg[(size_t)(i0 + mbase + r) * GGN + j0 + ncol] = acc[r] * scale;
  }
}

__global__ void global_softmax_kernel(const float* __restrict__ Sg,
                                      __bf16* __restrict__ Wg) {
  const int row = blockIdx.x, tid = threadIdx.x;
  const float* srow = Sg + (size_t)row * GGN;
  float vloc[13];
  float m = -3.402823e38f;
  int n = 0;
  for (int j = tid; j < GGN; j += 256) { float v = srow[j]; vloc[n++] = v; m = fmaxf(m, v); }
  __shared__ float red[256];
  red[tid] = m; __syncthreads();
  for (int off = 128; off > 0; off >>= 1) {
    if (tid < off) red[tid] = fmaxf(red[tid], red[tid + off]);
    __syncthreads();
  }
  m = red[0]; __syncthreads();
  float s = 0.f;
  for (int i = 0; i < n; ++i) { vloc[i] = __expf(vloc[i] - m); s += vloc[i]; }
  red[tid] = s; __syncthreads();
  for (int off = 128; off > 0; off >>= 1) {
    if (tid < off) red[tid] += red[tid + off];
    __syncthreads();
  }
  const float inv = 1.f / red[0];
  n = 0;
  for (int j = tid; j < GGN; j += 256) Wg[(size_t)row * GGN + j] = to_bf16(vloc[n++] * inv);
}

// Hg^T[i,c] = sum_j Wg[i,j] vg[c,j]; nearest-upsample x8 and add 0.25x into HIN.
// One wave handles both c-tiles (c 0..15 and 16..31), reusing the A tile.
__global__ void global_apply_kernel(const __bf16* __restrict__ Wg,
                                    const __bf16* __restrict__ vgT,
                                    float* __restrict__ HIN, int b) {
  const int lane = threadIdx.x;
  const int i0 = blockIdx.x * 16;
  v8f acc0 = {0.f,0.f,0.f,0.f,0.f,0.f,0.f,0.f};
  v8f acc1 = acc0;
  for (int kj = 0; kj < GGN / 32; ++kj) {
    v16bf a  = load_a16(Wg + (size_t)i0 * GGN + kj * 32, GGN, lane);
    const __bf16* Bb = vgT + ((size_t)b * GGN + (size_t)kj * 32) * GPAD;
    v16bf b0 = load_b16(Bb,      GPAD, lane);
    v16bf b1 = load_b16(Bb + 16, GPAD, lane);
    acc0 = wmma_bf16(a, b0, acc0);
    acc1 = wmma_bf16(a, b1, acc1);
  }
  const int ncol = lane & 15, mbase = (lane >> 4) * 8;
#pragma unroll
  for (int half = 0; half < 2; ++half) {
    const int c = half * 16 + ncol;
    if (c >= C) continue;
#pragma unroll
    for (int r = 0; r < 8; ++r) {
      const int g = i0 + mbase + r;
      const int gy = g / 56, gx = g % 56;
      const float val = 0.25f * (half ? acc1[r] : acc0[r]);
      float* base = HIN + (size_t)b * CHWn + (size_t)c * HW + (size_t)(gy * 8) * 448 + gx * 8;
      for (int dy = 0; dy < 8; ++dy)
#pragma unroll
        for (int dx = 0; dx < 8; ++dx) base[dy * 448 + dx] += val;
    }
  }
}

// out = x + proj_w @ HIN
__global__ void proj_kernel(const float* __restrict__ x, const float* __restrict__ pw,
                            const float* __restrict__ HIN, float* __restrict__ out) {
  __shared__ float s_pw[C * C];
  for (int i = threadIdx.x; i < C * C; i += blockDim.x) s_pw[i] = pw[i];
  __syncthreads();
  const int b  = blockIdx.y;
  const int hw = blockIdx.x * 256 + threadIdx.x;
  const float* Hb = HIN + (size_t)b * CHWn;
  const float* xb = x + (size_t)b * CHWn;
  float* ob = out + (size_t)b * CHWn;
  float h[C];
#pragma unroll
  for (int c = 0; c < C; ++c) h[c] = Hb[(size_t)c * HW + hw];
  for (int o = 0; o < C; ++o) {
    float a = 0.f;
#pragma unroll
    for (int c = 0; c < C; ++c) a += s_pw[o * C + c] * h[c];
    ob[(size_t)o * HW + hw] = xb[(size_t)o * HW + hw] + a;
  }
}

// ---------------- launch ----------------------------------------------------
extern "C" void kernel_launch(void* const* d_in, const int* in_sizes, int n_in,
                              void* d_out, int out_size, void* d_ws, size_t ws_size,
                              hipStream_t stream) {
  (void)in_sizes; (void)n_in; (void)out_size; (void)ws_size;
  const float* x    = (const float*)d_in[0];
  const float* gn_w = (const float*)d_in[1];
  const float* gn_b = (const float*)d_in[2];
  const float* q_w  = (const float*)d_in[3];
  const float* q_b  = (const float*)d_in[4];
  const float* k_w  = (const float*)d_in[5];
  const float* k_b  = (const float*)d_in[6];
  const float* v_w  = (const float*)d_in[7];
  const float* v_b  = (const float*)d_in[8];
  const float* p_w  = (const float*)d_in[9];
  float* out = (float*)d_out;

  // workspace layout (all chunks 16B-aligned)
  const size_t QT_B   = (size_t)B * NP * DP * 2;        // 49,545,216
  const size_t POOL_B = (size_t)B * GGN * GPAD * 2;     //    802,816
  char* p = (char*)d_ws;
  __bf16* Qt  = (__bf16*)p; p += QT_B;
  __bf16* Vt  = (__bf16*)p; p += QT_B;
  __bf16* Kb  = (__bf16*)p; p += QT_B;
  __bf16* qgT = (__bf16*)p; p += POOL_B;
  __bf16* kg  = (__bf16*)p; p += POOL_B;
  __bf16* vgT = (__bf16*)p; p += POOL_B;
  const size_t zero_bytes = (size_t)(p - (char*)d_ws);  // pad regions need zeros
  float*  Sp  = (float*)p;  p += (size_t)B * NP * NP * 4;
  __bf16* Wp  = (__bf16*)p; p += (size_t)B * NP * NP * 2;
  float*  Sg  = (float*)p;  p += (size_t)GGN * GGN * 4;    // reused per batch
  __bf16* Wg  = (__bf16*)p; p += (size_t)GGN * GGN * 2;    // reused per batch
  float*  HIN = (float*)p;  p += (size_t)B * CHWn * 4;
  float*  stats = (float*)p;

  zero_kernel<<<8192, 256, 0, stream>>>((uint4*)d_ws, zero_bytes / 16);
  gn_stats_kernel<<<B, 256, 0, stream>>>(x, stats);
  qkv_kernel<<<dim3(HW / 256, B), 256, 0, stream>>>(x, gn_w, gn_b, q_w, q_b, k_w,
                                                    k_b, v_w, v_b, stats, Qt, Kb, Vt);
  pool_kernel<<<(B * C * GGN) / 256, 256, 0, stream>>>(Qt, Kb, Vt, qgT, kg, vgT);

  patch_gram_kernel<<<dim3(7, 7, B), 32, 0, stream>>>(Qt, Kb, Sp);
  patch_softmax_kernel<<<dim3(NP, B), 256, 0, stream>>>(Sp, Wp);
  patch_apply_kernel<<<dim3(7, 216, B), 128, 0, stream>>>(Wp, Vt, HIN);

  for (int b = 0; b < B; ++b) {   // reuse Sg/Wg buffers; stream serializes
    global_gram_kernel<<<GGN / 16, 32, 0, stream>>>(qgT, kg, Sg, b);
    global_softmax_kernel<<<GGN, 256, 0, stream>>>(Sg, Wg);
    global_apply_kernel<<<GGN / 16, 32, 0, stream>>>(Wg, vgT, HIN, b);
  }

  proj_kernel<<<dim3(HW / 256, B), 256, 0, stream>>>(x, p_w, HIN, out);
}
